// style_mix_50148038148417
// MI455X (gfx1250) — matrix operand order repaired and checked by reading, not compiled, not run on hardware
//
#include <hip/hip_runtime.h>
#include <hip/hip_bf16.h>
#include <math.h>

typedef __attribute__((ext_vector_type(16))) __bf16       v16bf;
typedef __attribute__((ext_vector_type(8)))  float        v8f;
typedef __attribute__((ext_vector_type(4)))  float        f32x4;
typedef __attribute__((ext_vector_type(4)))  unsigned int u32x4;
typedef __attribute__((ext_vector_type(8)))  int          i32x8;
typedef __attribute__((ext_vector_type(4)))  int          i32x4;

#define BATCH 4
#define CH    512
#define CQD   128
#define NPOS  4096
#define OROWS 768   /* CQ + CQ + C */

union V16Cast { u32x4 u[2]; v16bf v; };
union PackB16 { __bf16 h[8]; u32x4 u; };

// Load a 16-bf16 WMMA fragment from two 16-byte-aligned 8-element runs.
static __device__ __forceinline__ v16bf load_frag_split(const __bf16* p0, const __bf16* p1) {
  V16Cast t;
  t.u[0] = *(const u32x4*)p0;
  t.u[1] = *(const u32x4*)p1;
  return t.v;
}
// Load a 16-bf16 WMMA fragment from one contiguous 32-byte run.
static __device__ __forceinline__ v16bf load_frag_contig(const __bf16* p) {
  return load_frag_split(p, p + 8);
}

// ---------------------------------------------------------------------------
// TDM: issue a 1-row tensor_load_to_lds of `bytes` contiguous bytes from
// `gaddr` into LDS byte offset `lds_off`.  D# built per CDNA5 ISA ch.8:
//  group0: count=1, lds_addr, global_addr[56:0], type=2
//  group1: data_size=1B, tensor_dim0=huge, tile_dim0=bytes, tile_dim1=1
// This toolchain's builtin takes 6 args (g0, g1, g2, g3, g4, cpol).
// ---------------------------------------------------------------------------
static __device__ __forceinline__ void tdm_load_1d(unsigned lds_off,
                                                   unsigned long long gaddr,
                                                   unsigned bytes) {
  u32x4 g0;
  g0[0] = 1u;                                   // count=1, user mode
  g0[1] = lds_off;                              // lds_addr (bytes)
  g0[2] = (unsigned)(gaddr & 0xFFFFFFFFull);    // global_addr[31:0]
  g0[3] = (unsigned)((gaddr >> 32) & 0x01FFFFFFull) | (2u << 30); // addr[56:32] | type=2

  i32x8 g1;
  g1[0] = 0;                                    // wg_mask=0, data_size=1B, no flags
  g1[1] = (int)0xFFFF0000u;                     // tensor_dim0 = 0x7FFFFFFF (lo 16 in [31:16])
  g1[2] = (int)0x00017FFFu;                     // tensor_dim0 hi | tensor_dim1=1
  g1[3] = (int)(bytes << 16);                   // tile_dim0 = bytes
  g1[4] = 1;                                    // tile_dim1 = 1, tile_dim2 = 0
  g1[5] = (int)bytes;                           // tensor_dim0_stride (don't care)
  g1[6] = 0;
  g1[7] = 0;

  i32x4 gz4 = {0, 0, 0, 0};
  i32x8 gz8 = {0, 0, 0, 0, 0, 0, 0, 0};
  __builtin_amdgcn_tensor_load_to_lds(g0, g1, gz4, gz4, gz8, 0);
}

// ---------------------------------------------------------------------------
// Kernel 1: fused q/k/v projection GEMM (bf16 WMMA, f32 accumulate).
// Out rows 0..127 = q (from vt), 128..255 = k (from vs), 256..767 = v (from vs).
// q stored transposed [B][N][128] (*attn_scale), k transposed [B][N][128],
// v natural [B][C][N] -- all bf16 in workspace.
// ---------------------------------------------------------------------------
__global__ __launch_bounds__(256)
void qkv_proj_kernel(const float* __restrict__ vt, const float* __restrict__ vs,
                     const float* __restrict__ Wq, const float* __restrict__ Wk,
                     const float* __restrict__ Wv,
                     __bf16* __restrict__ qws, __bf16* __restrict__ kws,
                     __bf16* __restrict__ vws)
{
  const int wave = threadIdx.x >> 5;
  const int lane = threadIdx.x & 31;
  const int hf   = lane >> 4;      // 16-lane half
  const int lr   = lane & 15;

  const int tiles_per_b = (OROWS / 16) * (NPOS / 16);   // 48 * 256 = 12288
  const int gw = blockIdx.x * 8 + wave;
  const int b  = gw / tiles_per_b;
  const int t  = gw % tiles_per_b;
  const int o_base = (t / (NPOS / 16)) * 16;
  const int n_base = (t % (NPOS / 16)) * 16;

  const float eq_scale   = 0.044194173824159216f;  // 1/sqrt(512)
  const float attn_scale = 0.044194173824159216f;  // 1/sqrt(512)

  const float* Wsrc; const float* xsrc; int region;
  if (o_base < CQD)            { Wsrc = Wq; xsrc = vt; region = 0; }
  else if (o_base < 2 * CQD)   { Wsrc = Wk; xsrc = vs; region = 1; }
  else                         { Wsrc = Wv; xsrc = vs; region = 2; }
  const int o_local_base = o_base - ((region == 0) ? 0 : (region == 1 ? CQD : 2 * CQD));

  const float* xb = xsrc + (size_t)b * CH * NPOS;

  const int a_row = o_local_base + lr;   // A row = output channel (lanes l, l+16 share row)
  const int ncol  = n_base + lr;         // B col = spatial position

  v8f acc = {};
  for (int kc = 0; kc < CH; kc += 32) {
    v16bf a, bop;
    const float* wrow = Wsrc + (size_t)a_row * CH + kc;
    #pragma unroll
    for (int e = 0; e < 8; ++e) {                 // A layout: e<8 -> K=hf*8+e, e>=8 -> K=16+hf*8+(e-8)
      a[e]     = (__bf16)(wrow[hf * 8 + e]      * eq_scale);
      a[e + 8] = (__bf16)(wrow[16 + hf * 8 + e] * eq_scale);
    }
    #pragma unroll
    for (int e = 0; e < 16; ++e) {                // B layout: e<8 -> K=hf*8+e, e>=8 -> K=16+hf*8+(e-8)
      bop[e] = (__bf16)(xb[(size_t)(kc + (e < 8 ? hf * 8 + e : 16 + hf * 8 + (e - 8))) * NPOS + ncol]);
    }
    acc = __builtin_amdgcn_wmma_f32_16x16x32_bf16(false, a, false, bop, (short)0, acc, false, false);
  }

  // C layout: VGPR i -> row M = i + 8*hf, col N = lr.  Region is wave-uniform.
  // q/k: rows i=0..7 are contiguous bf16 in [N][128] layout and 16B aligned
  //      -> pack into a single global_store_b128 per lane.
  if (region == 0) {
    PackB16 st;
    #pragma unroll
    for (int i = 0; i < 8; ++i) st.h[i] = (__bf16)(acc[i] * attn_scale);
    *(u32x4*)(qws + ((size_t)b * NPOS + ncol) * CQD + o_local_base + 8 * hf) = st.u;
  } else if (region == 1) {
    PackB16 st;
    #pragma unroll
    for (int i = 0; i < 8; ++i) st.h[i] = (__bf16)acc[i];
    *(u32x4*)(kws + ((size_t)b * NPOS + ncol) * CQD + o_local_base + 8 * hf) = st.u;
  } else {
    #pragma unroll
    for (int i = 0; i < 8; ++i)   // v layout [C][N]: rows stride N, scalar stores
      vws[((size_t)b * CH + o_local_base + i + 8 * hf) * NPOS + ncol] = (__bf16)acc[i];
  }
}

// ---------------------------------------------------------------------------
// Kernel 2: flash-attention. One block = (batch b, 16 query rows), 4 waves.
// Wave w owns output channels [128w, 128w+128) -> 8 f32 C fragments.
// The 32-key x 128-o K tile (8 KB, contiguous in k_ws) is staged into LDS by
// the Tensor Data Mover (wave 0), double-buffered, synced with
// s_wait_tensorcnt + workgroup barriers.  Each wave computes the shared
// 16x32 S tile from LDS (8 wmma, K=128), online softmax via shuffle
// reductions (C-fragment rows align with per-lane VGPR index), stages P
// through a private LDS tile (C-layout -> A-layout), then 8 PV wmma.
// ---------------------------------------------------------------------------
__global__ __launch_bounds__(128)
void flash_attn_kernel(const float* __restrict__ vt, const float* __restrict__ gamma_p,
                       const __bf16* __restrict__ qws, const __bf16* __restrict__ kws,
                       const __bf16* __restrict__ vws, float* __restrict__ out)
{
  __shared__ __align__(16) __bf16 kbuf[2][32 * CQD];  // 2 x 8 KB K tiles
  __shared__ __align__(16) __bf16 plds[4][16 * 32];   // per-wave P staging

  const int wave = threadIdx.x >> 5;
  const int lane = threadIdx.x & 31;
  const int hf   = lane >> 4;
  const int lr   = lane & 15;

  const int b      = blockIdx.x >> 8;          // 256 query tiles per batch
  const int n_base = (blockIdx.x & 255) * 16;
  const int cbase  = wave * 128;

  const __bf16* kbb = kws + (size_t)b * NPOS * CQD;
  const __bf16* vbb = vws + (size_t)b * CH * NPOS;

  const unsigned kbuf_off[2] = { (unsigned)(size_t)&kbuf[0][0],
                                 (unsigned)(size_t)&kbuf[1][0] };
  const unsigned ktile_bytes = 32 * CQD * 2;   // 8192

  // Prologue: DMA first K tile.
  if (wave == 0) {
    tdm_load_1d(kbuf_off[0], (unsigned long long)(size_t)kbb, ktile_bytes);
  }

  // Q A-fragments: 4 frags covering K(=o) 0..127. A layout split 8/8 per half.
  const __bf16* qrow = qws + ((size_t)b * NPOS + n_base + lr) * CQD;
  v16bf qa[4];
  #pragma unroll
  for (int f = 0; f < 4; ++f)
    qa[f] = load_frag_split(qrow + f * 32 + hf * 8, qrow + f * 32 + 16 + hf * 8);

  v8f acc[8];
  #pragma unroll
  for (int cb = 0; cb < 8; ++cb) acc[cb] = (v8f){};
  float rmax[8], rden[8];
  #pragma unroll
  for (int i = 0; i < 8; ++i) { rmax[i] = -3.0e38f; rden[i] = 0.0f; }

  __bf16* pw = &plds[wave][0];

  for (int m = 0; m < NPOS; m += 32) {
    const int cur = (m >> 5) & 1;
    // DMA next tile while waiting for the current one.
    if (wave == 0) {
      if (m + 32 < NPOS) {
        tdm_load_1d(kbuf_off[cur ^ 1],
                    (unsigned long long)(size_t)(kbb + (size_t)(m + 32) * CQD), ktile_bytes);
        __builtin_amdgcn_s_wait_tensorcnt(1);   // current tile done, next in flight
      } else {
        __builtin_amdgcn_s_wait_tensorcnt(0);   // last tile: drain
      }
    }
    __syncthreads();   // current K tile visible to all waves

    // ---- S = Q^T K for 32 keys (two 16x16 blocks), K-dim = 128, from LDS ----
    const __bf16* krow0 = &kbuf[cur][0] + (size_t)lr * CQD;         // keys m..m+15
    const __bf16* krow1 = &kbuf[cur][0] + (size_t)(16 + lr) * CQD;  // keys m+16..m+31
    v8f s0 = {}, s1 = {};
    #pragma unroll
    for (int f = 0; f < 4; ++f) {
      v16bf kb0 = load_frag_split(krow0 + f * 32 + hf * 8, krow0 + f * 32 + 16 + hf * 8);
      s0 = __builtin_amdgcn_wmma_f32_16x16x32_bf16(false, qa[f], false, kb0, (short)0, s0, false, false);
      v16bf kb1 = load_frag_split(krow1 + f * 32 + hf * 8, krow1 + f * 32 + 16 + hf * 8);
      s1 = __builtin_amdgcn_wmma_f32_16x16x32_bf16(false, qa[f], false, kb1, (short)0, s1, false, false);
    }

    // ---- online softmax; row M = i + 8*hf lives in VGPR i across 16 lanes ----
    float p0[8], p1[8], fs[8];
    #pragma unroll
    for (int i = 0; i < 8; ++i) {
      float mv = fmaxf(s0[i], s1[i]);
      mv = fmaxf(mv, __shfl_xor(mv, 1));
      mv = fmaxf(mv, __shfl_xor(mv, 2));
      mv = fmaxf(mv, __shfl_xor(mv, 4));
      mv = fmaxf(mv, __shfl_xor(mv, 8));
      const float nm = fmaxf(rmax[i], mv);
      p0[i] = __expf(s0[i] - nm);
      p1[i] = __expf(s1[i] - nm);
      float sv = p0[i] + p1[i];
      sv += __shfl_xor(sv, 1);
      sv += __shfl_xor(sv, 2);
      sv += __shfl_xor(sv, 4);
      sv += __shfl_xor(sv, 8);
      fs[i]   = __expf(rmax[i] - nm);
      rden[i] = rden[i] * fs[i] + sv;
      rmax[i] = nm;
    }
    #pragma unroll
    for (int cb = 0; cb < 8; ++cb)
      #pragma unroll
      for (int i = 0; i < 8; ++i) acc[cb][i] *= fs[i];

    // ---- P: C-layout -> LDS (16x32 bf16) -> A-layout fragment ----
    #pragma unroll
    for (int i = 0; i < 8; ++i) {
      const int row = i + 8 * hf;
      pw[row * 32 + lr]      = (__bf16)p0[i];
      pw[row * 32 + 16 + lr] = (__bf16)p1[i];
    }
    __builtin_amdgcn_wave_barrier();
    asm volatile("s_wait_dscnt 0" ::: "memory");   // intra-wave LDS RAW fence
    v16bf pa = load_frag_contig(pw + lr * 32 + hf * 16);

    // ---- acc += P * V^T : 8 channel blocks of 16, K = 32 keys ----
    const __bf16* vrow = vbb + (size_t)(cbase + lr) * NPOS + m + hf * 16;
    __builtin_prefetch((const void*)(vrow + 32), 0, 0);
    #pragma unroll
    for (int cb = 0; cb < 8; ++cb) {
      v16bf vb = load_frag_contig(vrow + (size_t)(cb * 16) * NPOS);
      acc[cb] = __builtin_amdgcn_wmma_f32_16x16x32_bf16(false, pa, false, vb, (short)0, acc[cb], false, false);
    }

    __syncthreads();   // all waves done with kbuf[cur] before wave 0 refills it
  }

  // ---- epilogue: out = gamma * (acc / denom) + vt ----
  // For each channel block, i=0..7 are consecutive n positions -> vectorize
  // the vt read + out write as 2 x b128 per lane.
  const float g = gamma_p[0];
  float invd[8];
  #pragma unroll
  for (int i = 0; i < 8; ++i) invd[i] = 1.0f / rden[i];

  #pragma unroll
  for (int cb = 0; cb < 8; ++cb) {
    const size_t base = ((size_t)b * CH + cbase + cb * 16 + lr) * NPOS + n_base + 8 * hf;
    f32x4 t0 = *(const f32x4*)(vt + base);
    f32x4 t1 = *(const f32x4*)(vt + base + 4);
    f32x4 o0, o1;
    #pragma unroll
    for (int j = 0; j < 4; ++j) {
      o0[j] = g * (acc[cb][j]     * invd[j])     + t0[j];
      o1[j] = g * (acc[cb][j + 4] * invd[j + 4]) + t1[j];
    }
    *(f32x4*)(out + base)     = o0;
    *(f32x4*)(out + base + 4) = o1;
  }
}

extern "C" void kernel_launch(void* const* d_in, const int* in_sizes, int n_in,
                              void* d_out, int out_size, void* d_ws, size_t ws_size,
                              hipStream_t stream) {
  (void)in_sizes; (void)n_in; (void)out_size; (void)ws_size;
  const float* vt    = (const float*)d_in[0];
  const float* vs    = (const float*)d_in[1];
  const float* Wq    = (const float*)d_in[2];
  const float* Wk    = (const float*)d_in[3];
  const float* Wv    = (const float*)d_in[4];
  const float* gamma = (const float*)d_in[5];
  float* out = (float*)d_out;

  // Workspace: q[B][N][128] bf16 (4 MB) | k[B][N][128] bf16 (4 MB) | v[B][C][N] bf16 (16 MB)
  __bf16* qws = (__bf16*)d_ws;
  __bf16* kws = qws + (size_t)BATCH * NPOS * CQD;
  __bf16* vws = kws + (size_t)BATCH * NPOS * CQD;

  // Kernel 1: B * (768/16)*(4096/16) = 49152 wave-tiles / 8 waves per block
  qkv_proj_kernel<<<dim3(6144), dim3(256), 0, stream>>>(vt, vs, Wq, Wk, Wv, qws, kws, vws);
  // Kernel 2: B * 256 query tiles, 4 waves each
  flash_attn_kernel<<<dim3(1024), dim3(128), 0, stream>>>(vt, gamma, qws, kws, vws, out);
}
